// TransformerHealingAgent_9096740733198
// MI455X (gfx1250) — compile-verified, matrix operand order checked by reading, not run on hardware
//
#include <hip/hip_runtime.h>
#include <hip/hip_bf16.h>
#include <math.h>

// ---------------------------------------------------------------------------
// TransformerHealingAgent forward on MI455X (gfx1250, wave32, WMMA + TDM).
// - Weights converted once per launch to transposed f16 (W^T[Nc][K]) so both
//   LDS panels are contiguous f16; fragments = 2x ds_load_b128, no cvt in the
//   hot loop.
// - Activations carried as f16 twins for GEMM A-operands; f32 masters kept
//   for residual/LN/softmax numerics.
// - GEMM: 8-wave workgroup, 128x128 tile, K-step 32, double-buffered LDS
//   panels staged by tensor_load_to_lds + s_wait_tensorcnt (DMA overlaps
//   WMMA). 8 v_wmma_f32_16x16x32_f16 per wave per K-step.
// - Fused all-pairs bond MLP never materializes the 268MB pair tensor.
// ---------------------------------------------------------------------------

typedef __attribute__((ext_vector_type(16))) _Float16 v16h;
typedef __attribute__((ext_vector_type(8)))  float    v8f;
typedef __attribute__((ext_vector_type(4)))  unsigned int u32x4;
typedef __attribute__((ext_vector_type(8)))  int          i32x8;
typedef __attribute__((ext_vector_type(4)))  int          i32x4;

#define DMODEL 512
#define NATOMS 256
#define BATCH  8
#define BN     (BATCH * NATOMS)   // 2048 rows

#if __has_builtin(__builtin_amdgcn_tensor_load_to_lds) && \
    __has_builtin(__builtin_amdgcn_s_wait_tensorcnt)
#define USE_TDM 1
#else
#define USE_TDM 0
#endif

// ---------------------------------------------------------------------------
// WMMA helpers (layouts per CDNA5 ISA 7.12.2, wave32)
// ---------------------------------------------------------------------------
__device__ __forceinline__ v8f wmma32(v16h a, v16h b, v8f c) {
  return __builtin_amdgcn_wmma_f32_16x16x32_f16(
      false, a, false, b, (short)0, c, false, false);
}

// Fragment from a row-major f16 matrix: lane&15 selects row (A) / col (B^T);
// two contiguous 8-half runs -> 2x ds/global b128.
__device__ __forceinline__ v16h frag_row16(const _Float16* Pm, int ld,
                                           int row_base, int col_base, int lane) {
  const int m  = lane & 15;
  const int kb = (lane >> 4) << 3;
  const _Float16* src = Pm + (size_t)(row_base + m) * ld + col_base + kb;
  v16h f;
#pragma unroll
  for (int j = 0; j < 8; ++j) { f[j] = src[j]; f[8 + j] = src[16 + j]; }
  return f;
}

// ---------------------------------------------------------------------------
// TDM: 2D f16 tile load global -> LDS (D# per CDNA5 ISA
// §8.3/8.4; group0: count=1, lds_addr, 57b global addr, type=2; group1:
// data_size=1 (2B), dims/tile/stride; groups2/3 zero for 2D).
// ---------------------------------------------------------------------------
__device__ __forceinline__ void tdm_load_2d16(const void* lds_ptr, const void* gptr,
                                              unsigned tensor_d0, unsigned tensor_d1,
                                              unsigned tile_d0, unsigned tile_d1,
                                              unsigned long long stride0_elems) {
#if USE_TDM
  const unsigned long long ga = (unsigned long long)gptr;
  u32x4 g0;
  g0[0] = 1u;                                                // count=1
  g0[1] = (unsigned)(unsigned long long)lds_ptr;             // LDS byte address
  g0[2] = (unsigned)(ga & 0xFFFFFFFFu);
  g0[3] = (unsigned)((ga >> 32) & 0x01FFFFFFu) | (2u << 30); // addr[56:32] | type=2
  i32x8 g1;
  g1[0] = (int)0x00010000u;                                  // data_size=1 (2 bytes)
  g1[1] = (int)((tensor_d0 & 0xFFFFu) << 16);
  g1[2] = (int)(((tensor_d0 >> 16) & 0xFFFFu) | ((tensor_d1 & 0xFFFFu) << 16));
  g1[3] = (int)(((tensor_d1 >> 16) & 0xFFFFu) | ((tile_d0 & 0xFFFFu) << 16));
  g1[4] = (int)(tile_d1 & 0xFFFFu);
  g1[5] = (int)(unsigned)(stride0_elems & 0xFFFFFFFFu);
  g1[6] = (int)(unsigned)((stride0_elems >> 32) & 0xFFFFu);
  g1[7] = 0;
  i32x4 gz = {0, 0, 0, 0};
#if defined(__clang_major__) && (__clang_major__ >= 23)
  i32x8 gz8 = {0, 0, 0, 0, 0, 0, 0, 0};
  __builtin_amdgcn_tensor_load_to_lds(g0, g1, gz, gz, gz8, 0);
#else
  __builtin_amdgcn_tensor_load_to_lds(g0, g1, gz, gz, 0);
#endif
#else
  (void)lds_ptr; (void)gptr; (void)tensor_d0; (void)tensor_d1;
  (void)tile_d0; (void)tile_d1; (void)stride0_elems;
#endif
}

// ---------------------------------------------------------------------------
// Weight prep: W[K][Nc] f32 -> W^T[Nc][K] f16.  One-time pass per launch.
// ---------------------------------------------------------------------------
__global__ void k_w16t(const float* __restrict__ W, _Float16* __restrict__ Wt,
                       int K, int Nc) {
  const int idx = blockIdx.x * 256 + threadIdx.x;   // over Nc*K, k fastest
  const int n = idx / K, k = idx - n * K;
  Wt[idx] = (_Float16)W[(size_t)k * Nc + n];
}

// ---------------------------------------------------------------------------
// Blocked GEMM: C = act(A16[M,K] @ Wt^T + bias).  A16 f16 row-major,
// Wt f16 [Nc][K].  8 waves, 128x128 tile, double-buffered 128x32 f16 panels
// staged by TDM.  Outputs: C (f32, optional) and C16 (f16 twin, optional).
// ---------------------------------------------------------------------------
__global__ void k_gemm(const _Float16* __restrict__ A16, const _Float16* __restrict__ Wt,
                       const float* __restrict__ bias, float* __restrict__ C,
                       _Float16* __restrict__ C16, int M, int K, int Nc, int act) {
  __shared__ _Float16 sA[2][128 * 32];   // [row][k]
  __shared__ _Float16 sB[2][128 * 32];   // [col][k]
  const int lane = threadIdx.x & 31;
  const int wave = threadIdx.x >> 5;
  const int wm = wave >> 2;                  // 0..1
  const int wn = wave & 3;                   // 0..3
  const int row0 = blockIdx.y * 128;
  const int col0 = blockIdx.x * 128;
  const int nsteps = K >> 5;

  v8f acc[4][2];
#pragma unroll
  for (int mt = 0; mt < 4; ++mt)
#pragma unroll
    for (int nt = 0; nt < 2; ++nt) acc[mt][nt] = (v8f){};

#if USE_TDM
  if (wave == 0) {
    tdm_load_2d16(&sA[0][0], A16 + (size_t)row0 * K, (unsigned)K, (unsigned)M,
                  32u, 128u, (unsigned long long)K);
    tdm_load_2d16(&sB[0][0], Wt + (size_t)col0 * K, (unsigned)K, (unsigned)Nc,
                  32u, 128u, (unsigned long long)K);
    __builtin_amdgcn_s_wait_tensorcnt(0);
  }
#else
  for (int idx = threadIdx.x; idx < 512; idx += 256) {
    const int r = idx >> 2, c = (idx & 3) << 3;
    *(float4*)&sA[0][r * 32 + c] = *(const float4*)(A16 + (size_t)(row0 + r) * K + c);
    *(float4*)&sB[0][r * 32 + c] = *(const float4*)(Wt  + (size_t)(col0 + r) * K + c);
  }
#endif
  __syncthreads();

  for (int s = 0; s < nsteps; ++s) {
    const int cur = s & 1;
#if USE_TDM
    if (wave == 0 && s + 1 < nsteps) {       // prefetch next panels (async DMA)
      const int k1 = (s + 1) << 5;
      tdm_load_2d16(&sA[cur ^ 1][0], A16 + (size_t)row0 * K + k1, (unsigned)K,
                    (unsigned)M, 32u, 128u, (unsigned long long)K);
      tdm_load_2d16(&sB[cur ^ 1][0], Wt + (size_t)col0 * K + k1, (unsigned)K,
                    (unsigned)Nc, 32u, 128u, (unsigned long long)K);
    }
#endif
    v16h af[4];
#pragma unroll
    for (int mt = 0; mt < 4; ++mt)
      af[mt] = frag_row16(&sA[cur][0], 32, wm * 64 + mt * 16, 0, lane);
    v16h bf[2];
#pragma unroll
    for (int nt = 0; nt < 2; ++nt)
      bf[nt] = frag_row16(&sB[cur][0], 32, wn * 32 + nt * 16, 0, lane);
#pragma unroll
    for (int mt = 0; mt < 4; ++mt)
#pragma unroll
      for (int nt = 0; nt < 2; ++nt)
        acc[mt][nt] = wmma32(af[mt], bf[nt], acc[mt][nt]);

    __syncthreads();                          // all waves done reading cur
#if USE_TDM
    if (wave == 0 && s + 1 < nsteps) __builtin_amdgcn_s_wait_tensorcnt(0);
#else
    if (s + 1 < nsteps) {
      const int k1 = (s + 1) << 5;
      for (int idx = threadIdx.x; idx < 512; idx += 256) {
        const int r = idx >> 2, c = (idx & 3) << 3;
        *(float4*)&sA[cur ^ 1][r * 32 + c] =
            *(const float4*)(A16 + (size_t)(row0 + r) * K + k1 + c);
        *(float4*)&sB[cur ^ 1][r * 32 + c] =
            *(const float4*)(Wt + (size_t)(col0 + r) * K + k1 + c);
      }
    }
#endif
    __syncthreads();                          // next panels published
  }

  const int n  = lane & 15;
  const int mb = (lane >> 4) << 3;
#pragma unroll
  for (int mt = 0; mt < 4; ++mt) {
#pragma unroll
    for (int nt = 0; nt < 2; ++nt) {
      const int cg = col0 + wn * 32 + nt * 16 + n;
      const int rg = row0 + wm * 64 + mt * 16 + mb;
      const float bv = bias ? bias[cg] : 0.0f;
#pragma unroll
      for (int r = 0; r < 8; ++r) {
        float v = acc[mt][nt][r] + bv;
        if (act == 1) v = fmaxf(v, 0.0f);
        if (C)   C[(size_t)(rg + r) * Nc + cg]   = v;
        if (C16) C16[(size_t)(rg + r) * Nc + cg] = (_Float16)v;
      }
    }
  }
}

// ---------------------------------------------------------------------------
// Input projection (19 -> 512) + sinusoidal PE; writes f32 + f16 twin.
// ---------------------------------------------------------------------------
__global__ void k_input_pe(const float* __restrict__ x, const float* __restrict__ w,
                           const float* __restrict__ b, float* __restrict__ out,
                           _Float16* __restrict__ out16) {
  const int row = blockIdx.x;
  const int n   = row & (NATOMS - 1);
  const float* xi = x + (size_t)row * 19;
  for (int d = threadIdx.x; d < DMODEL; d += blockDim.x) {
    float s = b[d];
#pragma unroll
    for (int k = 0; k < 19; ++k) s += xi[k] * w[k * DMODEL + d];
    const float div = __expf(-(float)(d & ~1) * (9.21034037198f / (float)DMODEL));
    const float ang = (float)n * div;
    s += (d & 1) ? __cosf(ang) : __sinf(ang);
    out[(size_t)row * DMODEL + d]   = s;
    out16[(size_t)row * DMODEL + d] = (_Float16)s;
  }
}

// ---------------------------------------------------------------------------
// out/out16 = LayerNorm(X + Y) * g + b.  One wave per row of 512.
// ---------------------------------------------------------------------------
__global__ void k_ln_residual(const float* __restrict__ X, const float* __restrict__ Y,
                              const float* __restrict__ g, const float* __restrict__ bt,
                              float* __restrict__ Out, _Float16* __restrict__ Out16) {
  const int row = blockIdx.x, lane = threadIdx.x;
  const float* xr = X + (size_t)row * DMODEL;
  const float* yr = Y + (size_t)row * DMODEL;
  float vals[DMODEL / 32];
  float s = 0.0f;
#pragma unroll
  for (int i = 0; i < DMODEL / 32; ++i) {
    float v = xr[lane + i * 32] + yr[lane + i * 32];
    vals[i] = v;  s += v;
  }
#pragma unroll
  for (int m = 16; m >= 1; m >>= 1) s += __shfl_xor(s, m, 32);
  const float mean = s * (1.0f / DMODEL);
  float vs = 0.0f;
#pragma unroll
  for (int i = 0; i < DMODEL / 32; ++i) { float d = vals[i] - mean; vs += d * d; }
#pragma unroll
  for (int m = 16; m >= 1; m >>= 1) vs += __shfl_xor(vs, m, 32);
  const float rstd = rsqrtf(vs * (1.0f / DMODEL) + 1e-5f);
#pragma unroll
  for (int i = 0; i < DMODEL / 32; ++i) {
    const float v = (vals[i] - mean) * rstd * g[lane + i * 32] + bt[lane + i * 32];
    Out[(size_t)row * DMODEL + lane + i * 32]   = v;
    Out16[(size_t)row * DMODEL + lane + i * 32] = (_Float16)v;
  }
}

// ---------------------------------------------------------------------------
// Attention on f16 qkv: grid (N/16, H, B), one wave per 16 query rows.
// Q/K fragments are contiguous f16 runs; scores f32 in LDS; P@V via WMMA.
// Writes f16 output twin (only consumed by the next GEMM).
// ---------------------------------------------------------------------------
__global__ void k_attention(const _Float16* __restrict__ qkv, _Float16* __restrict__ out16,
                            int hd) {
  const int lane = threadIdx.x;
  const int qt = blockIdx.x, h = blockIdx.y, b = blockIdx.z;
  const int ld = 3 * DMODEL;
  const _Float16* Qb = qkv + (size_t)(b * NATOMS) * ld + h * hd;
  const _Float16* Kb = Qb + DMODEL;
  const _Float16* Vb = Qb + 2 * DMODEL;
  __shared__ float sc[16 * NATOMS];                 // 16 KB
  const float scale = rsqrtf((float)hd);
  const int n  = lane & 15;
  const int kb = (lane >> 4) << 3;
  const int mb = (lane >> 4) << 3;

  // S = Q @ K^T  (col n of K^T = row n of K -> contiguous f16 run)
  for (int jt = 0; jt < NATOMS / 16; ++jt) {
    v8f acc = {};
    for (int k0 = 0; k0 < hd; k0 += 32) {
      v16h a  = frag_row16(Qb, ld, qt * 16, k0, lane);
      v16h bf = frag_row16(Kb, ld, jt * 16, k0, lane);
      acc = wmma32(a, bf, acc);
    }
#pragma unroll
    for (int r = 0; r < 8; ++r) sc[(mb + r) * NATOMS + jt * 16 + n] = acc[r] * scale;
  }
  __syncthreads();

  if (lane < 16) {
    float* rowp = sc + lane * NATOMS;
    float mx = -1e30f;
    for (int j = 0; j < NATOMS; ++j) mx = fmaxf(mx, rowp[j]);
    float sum = 0.0f;
    for (int j = 0; j < NATOMS; ++j) { float e = __expf(rowp[j] - mx); rowp[j] = e; sum += e; }
    const float inv = 1.0f / sum;
    for (int j = 0; j < NATOMS; ++j) rowp[j] *= inv;
  }
  __syncthreads();

  // O = P @ V
  for (int nt = 0; nt < hd / 16; ++nt) {
    v8f acc = {};
    for (int k0 = 0; k0 < NATOMS; k0 += 32) {
      v16h a;
      const float* srcp = sc + (lane & 15) * NATOMS + k0 + kb;
#pragma unroll
      for (int j = 0; j < 8; ++j) { a[j] = (_Float16)srcp[j]; a[8 + j] = (_Float16)srcp[16 + j]; }
      v16h bf;
      const _Float16* srcv = Vb + (size_t)(k0 + kb) * ld + nt * 16 + n;
#pragma unroll
      for (int j = 0; j < 8; ++j) {
        bf[j]     = srcv[(size_t)j * ld];
        bf[8 + j] = srcv[(size_t)(16 + j) * ld];
      }
      acc = wmma32(a, bf, acc);
    }
#pragma unroll
    for (int r = 0; r < 8; ++r)
      out16[(size_t)(b * NATOMS + qt * 16 + mb + r) * DMODEL + h * hd + nt * 16 + n] =
          (_Float16)acc[r];
  }
}

// ---------------------------------------------------------------------------
// Fused all-pairs bond MLP. One wave = 16 pairs (fixed i, 16 consecutive j).
// A = relu(hi[i]+hj[j]) built in-register (f32 adds, then cvt); W2^T f16 gives
// contiguous B fragments. 64 WMMAs/wave; bond3 + sigmoid fused via shfl.
// ---------------------------------------------------------------------------
__global__ void k_bond_pairs(const float* __restrict__ hi, const float* __restrict__ hj,
                             const _Float16* __restrict__ w2t, const float* __restrict__ b2,
                             const float* __restrict__ w3, const float* __restrict__ b3,
                             float* __restrict__ bond) {
  const int lane = threadIdx.x;
  const int b  = blockIdx.y;
  const int m0 = blockIdx.x * 16;
  const int i  = m0 >> 8;
  const int j0 = m0 & (NATOMS - 1);
  const float* hir = hi + ((size_t)b * NATOMS + i)  * 256;
  const float* hjb = hj + ((size_t)b * NATOMS + j0) * 256;
  const int mrow = lane & 15;
  const int kb   = (lane >> 4) << 3;

  v8f acc[8];
#pragma unroll
  for (int nt = 0; nt < 8; ++nt) acc[nt] = (v8f){};

  for (int k0 = 0; k0 < 256; k0 += 32) {
    v16h a;
    const float* hjr = hjb + (size_t)mrow * 256 + k0 + kb;
    const float* hic = hir + k0 + kb;
#pragma unroll
    for (int j = 0; j < 8; ++j) {
      a[j]     = (_Float16)fmaxf(hic[j]      + hjr[j],      0.0f);
      a[8 + j] = (_Float16)fmaxf(hic[16 + j] + hjr[16 + j], 0.0f);
    }
#pragma unroll
    for (int nt = 0; nt < 8; ++nt) {
      v16h bf = frag_row16(w2t, 256, nt * 16, k0, lane);
      acc[nt] = wmma32(a, bf, acc[nt]);
    }
  }

  const int n  = lane & 15;
  const int mb = (lane >> 4) << 3;
  float part[8];
#pragma unroll
  for (int r = 0; r < 8; ++r) part[r] = 0.0f;
#pragma unroll
  for (int nt = 0; nt < 8; ++nt) {
    const int col = nt * 16 + n;
    const float bb = b2[col], ww = w3[col];
#pragma unroll
    for (int r = 0; r < 8; ++r) part[r] += fmaxf(acc[nt][r] + bb, 0.0f) * ww;
  }
#pragma unroll
  for (int r = 0; r < 8; ++r) {
    float s = part[r];
#pragma unroll
    for (int msk = 1; msk < 16; msk <<= 1) s += __shfl_xor(s, msk, 32);
    part[r] = s;
  }
  if ((lane & 15) == 0) {
    const float b3v = b3[0];
#pragma unroll
    for (int r = 0; r < 8; ++r) {
      const int m = m0 + mb + r;
      bond[(size_t)b * NATOMS * NATOMS + m] = 1.0f / (1.0f + __expf(-(part[r] + b3v)));
    }
  }
}

// ---------------------------------------------------------------------------
// Tiny heads (256 -> 3 / 10 / 1), one wave per row.
// ---------------------------------------------------------------------------
__global__ void k_pos_head(const float* __restrict__ hb, const float* __restrict__ w,
                           const float* __restrict__ b, float* __restrict__ out) {
  const int row = blockIdx.x, lane = threadIdx.x;
  if (lane < 3) {
    const float* hr = hb + (size_t)row * 256;
    float s = b[lane];
    for (int k = 0; k < 256; ++k) s += hr[k] * w[k * 3 + lane];
    out[(size_t)row * 3 + lane] = tanhf(s) * 0.1f;
  }
}

__global__ void k_typ_head(const float* __restrict__ hb, const float* __restrict__ w,
                           const float* __restrict__ b, float* __restrict__ out) {
  const int row = blockIdx.x, lane = threadIdx.x;
  __shared__ float lg[10];
  const float* hr = hb + (size_t)row * 256;
  if (lane < 10) {
    float s = b[lane];
    for (int k = 0; k < 256; ++k) s += hr[k] * w[k * 10 + lane];
    lg[lane] = s;
  }
  __syncthreads();
  if (lane == 0) {
    float mx = -1e30f;
    for (int o = 0; o < 10; ++o) mx = fmaxf(mx, lg[o]);
    float e[10], sum = 0.0f;
    for (int o = 0; o < 10; ++o) { e[o] = __expf(lg[o] - mx); sum += e[o]; }
    const float inv = 1.0f / sum;
    for (int o = 0; o < 10; ++o) out[(size_t)row * 10 + o] = e[o] * inv;
  }
}

__global__ void k_cls_head(const float* __restrict__ hb, const float* __restrict__ w,
                           const float* __restrict__ b, float* __restrict__ out) {
  const int row = blockIdx.x, lane = threadIdx.x;
  if (lane == 0) {
    const float* hr = hb + (size_t)row * 256;
    float s = b[0];
    for (int k = 0; k < 256; ++k) s += hr[k] * w[k];
    out[row] = 1.0f / (1.0f + __expf(-s));
  }
}

// ---------------------------------------------------------------------------
// Host orchestration.  Input flattening assumed jax-pytree order (dict keys
// sorted at each level): params first (bond1,bond2,bond3,cls1,cls2,dout,dqkv,
// inp,layers,pos1,pos2,typ1,typ2; each lin dict b then w; layers[l] keys
// ff1,ff2,ln1b,ln1g,ln2b,ln2g,out,qkv), then x at index 96.
// ---------------------------------------------------------------------------
extern "C" void kernel_launch(void* const* d_in, const int* in_sizes, int n_in,
                              void* d_out, int out_size, void* d_ws, size_t ws_size,
                              hipStream_t stream) {
  (void)in_sizes; (void)n_in; (void)out_size; (void)ws_size;
  auto P = [&](int i) -> const float* { return (const float*)d_in[i]; };

  const float* bond1_b = P(0);  const float* bond1_w = P(1);   // [1024,256]
  const float* bond2_b = P(2);  const float* bond2_w = P(3);   // [256,128]
  const float* bond3_b = P(4);  const float* bond3_w = P(5);   // [128,1]
  const float* cls1_b  = P(6);  const float* cls1_w  = P(7);   // [512,256]
  const float* cls2_b  = P(8);  const float* cls2_w  = P(9);   // [256,1]
  const float* dout_b  = P(10); const float* dout_w  = P(11);  // [512,512]
  const float* dqkv_b  = P(12); const float* dqkv_w  = P(13);  // [512,1536]
  const float* inp_b   = P(14); const float* inp_w   = P(15);  // [19,512]
  const float* pos1_b = P(88); const float* pos1_w = P(89);
  const float* pos2_b = P(90); const float* pos2_w = P(91);
  const float* typ1_b = P(92); const float* typ1_w = P(93);
  const float* typ2_b = P(94); const float* typ2_w = P(95);
  const float* x      = P(96);                                 // [8,256,19]

  // ---- workspace bump allocator (256B aligned) ----
  char* wp = (char*)d_ws;
  auto alloc = [&](size_t bytes) -> void* {
    void* r = (void*)wp; wp += (bytes + 255) & ~(size_t)255; return r;
  };
  float*    h     = (float*)   alloc((size_t)BN * DMODEL * 4);
  float*    h2    = (float*)   alloc((size_t)BN * DMODEL * 4);
  float*    hi    = (float*)   alloc((size_t)BN * 256 * 4);
  float*    hj    = (float*)   alloc((size_t)BN * 256 * 4);
  float*    hb    = (float*)   alloc((size_t)BN * 256 * 4);
  _Float16* h16   = (_Float16*)alloc((size_t)BN * DMODEL * 2);
  _Float16* h2_16 = (_Float16*)alloc((size_t)BN * DMODEL * 2);
  _Float16* qkv16 = (_Float16*)alloc((size_t)BN * 3 * DMODEL * 2);
  _Float16* att16 = (_Float16*)alloc((size_t)BN * DMODEL * 2);
  _Float16* ff16  = (_Float16*)alloc((size_t)BN * 4 * DMODEL * 2);

  // transposed f16 weights
  auto prep = [&](const float* W, int K, int Nc) -> _Float16* {
    _Float16* Wt = (_Float16*)alloc((size_t)K * Nc * 2);
    k_w16t<<<(K * Nc) / 256, 256, 0, stream>>>(W, Wt, K, Nc);
    return Wt;
  };
  _Float16 *WqT[6], *WoT[6], *Wf1T[6], *Wf2T[6];
  for (int l = 0; l < 6; ++l) {
    const int base = 16 + l * 12;
    Wf1T[l] = prep(P(base + 1), DMODEL, 4 * DMODEL);
    Wf2T[l] = prep(P(base + 3), 4 * DMODEL, DMODEL);
    WoT[l]  = prep(P(base + 9), DMODEL, DMODEL);
    WqT[l]  = prep(P(base + 11), DMODEL, 3 * DMODEL);
  }
  _Float16* dqkvT = prep(dqkv_w, DMODEL, 3 * DMODEL);
  _Float16* doutT = prep(dout_w, DMODEL, DMODEL);
  _Float16* cls1T = prep(cls1_w, DMODEL, 256);
  _Float16* pos1T = prep(pos1_w, DMODEL, 256);
  _Float16* typ1T = prep(typ1_w, DMODEL, 256);
  _Float16* b1aT  = prep(bond1_w,             DMODEL, 256);   // W1[:512]
  _Float16* b1bT  = prep(bond1_w + 512 * 256, DMODEL, 256);   // W1[512:]
  _Float16* w2T   = prep(bond2_w, 256, 128);                  // -> [128][256]

  float* out       = (float*)d_out;
  float* out_bond  = out;
  float* out_pos   = out_bond + (size_t)BATCH * NATOMS * NATOMS;
  float* out_typ   = out_pos  + (size_t)BATCH * NATOMS * 3;
  float* out_dmg   = out_typ  + (size_t)BATCH * NATOMS * 10;

  const dim3 wave(32);
  auto gemm = [&](const _Float16* A16, const _Float16* Wt, const float* bias,
                  float* C, _Float16* C16, int M, int K, int Nc, int act) {
    k_gemm<<<dim3(Nc / 128, M / 128), dim3(256), 0, stream>>>(
        A16, Wt, bias, C, C16, M, K, Nc, act);
  };

  // input projection + positional encoding
  k_input_pe<<<BN, 128, 0, stream>>>(x, inp_w, inp_b, h, h16);

  // encoder layers
  for (int l = 0; l < 6; ++l) {
    const int base = 16 + l * 12;
    const float* ff1b = P(base + 0), *ff2b = P(base + 2);
    const float* ln1b = P(base + 4), *ln1g = P(base + 5);
    const float* ln2b = P(base + 6), *ln2g = P(base + 7);
    const float* outb = P(base + 8), *qkvb = P(base + 10);

    gemm(h16, WqT[l], qkvb, nullptr, qkv16, BN, DMODEL, 3 * DMODEL, 0);
    k_attention<<<dim3(NATOMS / 16, 8, BATCH), wave, 0, stream>>>(qkv16, att16, 64);
    gemm(att16, WoT[l], outb, h2, nullptr, BN, DMODEL, DMODEL, 0);
    k_ln_residual<<<BN, wave, 0, stream>>>(h, h2, ln1g, ln1b, h, h16);
    gemm(h16, Wf1T[l], ff1b, nullptr, ff16, BN, DMODEL, 4 * DMODEL, 1);
    gemm(ff16, Wf2T[l], ff2b, h2, nullptr, BN, 4 * DMODEL, DMODEL, 0);
    k_ln_residual<<<BN, wave, 0, stream>>>(h, h2, ln2g, ln2b, h, h16);
  }

  // all-pairs bond head
  gemm(h16, b1aT, nullptr, hi, nullptr, BN, DMODEL, 256, 0);
  gemm(h16, b1bT, bond1_b, hj, nullptr, BN, DMODEL, 256, 0);
  k_bond_pairs<<<dim3(NATOMS * NATOMS / 16, BATCH), wave, 0, stream>>>(
      hi, hj, w2T, bond2_b, bond3_w, bond3_b, out_bond);

  // position / type heads
  gemm(h16, pos1T, pos1_b, hb, nullptr, BN, DMODEL, 256, 1);
  k_pos_head<<<BN, wave, 0, stream>>>(hb, pos2_w, pos2_b, out_pos);
  gemm(h16, typ1T, typ1_b, hb, nullptr, BN, DMODEL, 256, 1);
  k_typ_head<<<BN, wave, 0, stream>>>(hb, typ2_w, typ2_b, out_typ);

  // damage head: 4-head MHA (hd=128) -> out proj -> cls MLP -> sigmoid
  gemm(h16, dqkvT, dqkv_b, nullptr, qkv16, BN, DMODEL, 3 * DMODEL, 0);
  k_attention<<<dim3(NATOMS / 16, 4, BATCH), wave, 0, stream>>>(qkv16, att16, 128);
  gemm(att16, doutT, dout_b, nullptr, h2_16, BN, DMODEL, DMODEL, 0);
  gemm(h2_16, cls1T, cls1_b, hb, nullptr, BN, DMODEL, 256, 1);
  k_cls_head<<<BN, wave, 0, stream>>>(hb, cls2_w, cls2_b, out_dmg);
}
